// GCN_16801912062630
// MI455X (gfx1250) — matrix operand order repaired and checked by reading, not compiled, not run on hardware
//
#include <hip/hip_runtime.h>

#define N_NODES 10000
#define E_EDGES 640000
#define C_IN    256
#define C_HID   128
#define C_OUT   64

typedef float    v2f  __attribute__((ext_vector_type(2)));
typedef float    v8f  __attribute__((ext_vector_type(8)));
typedef _Float16 v16h __attribute__((ext_vector_type(16)));

#if defined(__has_builtin)
#if __has_builtin(__builtin_amdgcn_wmma_f32_16x16x4_f32)
#define USE_WMMA_F32 1
#endif
#endif

// ---------------- degree / normalization ----------------
__global__ void deg_init_k(float* __restrict__ dinv, int n) {
  int i = blockIdx.x * blockDim.x + threadIdx.x;
  if (i < n) dinv[i] = 1.0f;  // self-loop contributes 1 to every degree
}

__global__ void deg_count_k(float* __restrict__ dinv, const int* __restrict__ dst, int e) {
  int i = blockIdx.x * blockDim.x + threadIdx.x;
  if (i < e)
    __hip_atomic_fetch_add(&dinv[dst[i]], 1.0f, __ATOMIC_RELAXED, __HIP_MEMORY_SCOPE_AGENT);
}

__global__ void rsqrt_k(float* __restrict__ dinv, int n) {
  int i = blockIdx.x * blockDim.x + threadIdx.x;
  if (i < n) dinv[i] = rsqrtf(dinv[i]);  // deg >= 1 always
}

// ---------------- WMMA GEMM: D[M,Ncols] = A[M,K] @ B[K,Ncols] ----------------
// One wave per 16x16 output tile. blockDim = 128 (4 waves); wave w handles
// col-tile blockIdx.y*4 + w. M tiles on blockIdx.x. M % 16 == 0, Ncols % 64 == 0,
// K % 32 == 0 for all call sites, so no masking: EXEC is all-1s (WMMA requirement).
__global__ __launch_bounds__(128)
void gemm_wmma_k(const float* __restrict__ A, const float* __restrict__ B,
                 float* __restrict__ D, int K, int Ncols) {
  const int lane = threadIdx.x & 31;
  const int wave = threadIdx.x >> 5;
  const int row0 = blockIdx.x * 16;
  const int col0 = (blockIdx.y * 4 + wave) * 16;
  const int r    = lane & 15;   // A row within tile / B,D column within tile
  const int half = lane >> 4;   // K-half selector (ISA layout)

  const float* Arow = A + (size_t)(row0 + r) * K;
  const float* Bcol = B + col0 + r;
  v8f acc = {};

#if defined(USE_WMMA_F32)
  // V_WMMA_F32_16X16X4_F32: full fp32 precision.
  // A 16x4: lanes0-15 -> K={0,1}, lanes16-31 -> K={2,3} (VGPR0,VGPR1)
  for (int k = 0; k < K; k += 4) {
    const int kb = k + half * 2;
    v2f a = *(const v2f*)(Arow + kb);              // 8B-aligned (kb even)
    v2f b;
    b.x = Bcol[(size_t)kb * Ncols];
    b.y = Bcol[(size_t)(kb + 1) * Ncols];
    acc = __builtin_amdgcn_wmma_f32_16x16x4_f32(false, a, false, b,
                                                (short)0, acc, false, false);
  }
#else
  // Fallback: codegen-confirmed V_WMMA_F32_16X16X32_F16 (fp16 in, fp32 accum).
  for (int k = 0; k < K; k += 32) {
    v16h a, b;
#pragma unroll
    for (int rr = 0; rr < 8; ++rr) {
      const int ka = k + half * 16 + 2 * rr;
      a[2 * rr]     = (_Float16)Arow[ka];
      a[2 * rr + 1] = (_Float16)Arow[ka + 1];
      const int kb = k + half * 16 + 2 * rr;
      b[2 * rr]     = (_Float16)Bcol[(size_t)kb * Ncols];
      b[2 * rr + 1] = (_Float16)Bcol[(size_t)(kb + 1) * Ncols];
    }
    acc = __builtin_amdgcn_wmma_f32_16x16x32_f16(false, a, false, b,
                                                 (short)0, acc, false, false);
  }
#endif

  // C/D layout: VGPR i -> row i + 8*half, lane%16 -> column.
  float* Drow = D + (size_t)(row0 + half * 8) * Ncols + col0 + r;
#pragma unroll
  for (int i = 0; i < 8; ++i) Drow[(size_t)i * Ncols] = acc[i];
}

// ---------------- aggregation ----------------
// out = bias + dinv[i]^2 * h[i]  (self-loop term + bias, no atomics needed)
template <int C>
__global__ void init_out_k(const float* __restrict__ h, const float* __restrict__ dinv,
                           const float* __restrict__ bias, float* __restrict__ out, int n) {
  int t = blockIdx.x * blockDim.x + threadIdx.x;
  if (t >= n * C) return;
  int i = t / C;
  int f = t - i * C;
  float di = dinv[i];
  out[t] = bias[f] + di * di * h[t];
}

// out[dst] += dinv[src]*dinv[dst] * h[src], 4 features per thread (float4 gather)
template <int C>
__global__ void scatter_k(const float* __restrict__ h, const float* __restrict__ dinv,
                          const int* __restrict__ src, const int* __restrict__ dst,
                          float* __restrict__ out, int e) {
  constexpr int CH = C / 4;
  int t = blockIdx.x * blockDim.x + threadIdx.x;
  if (t >= e * CH) return;
  int ei = t / CH;
  int fc = (t - ei * CH) * 4;
  int s = src[ei], d = dst[ei];
  float w = dinv[s] * dinv[d];
  const float4 hv = *(const float4*)(h + (size_t)s * C + fc);
  float* o = out + (size_t)d * C + fc;
  __hip_atomic_fetch_add(o + 0, w * hv.x, __ATOMIC_RELAXED, __HIP_MEMORY_SCOPE_AGENT);
  __hip_atomic_fetch_add(o + 1, w * hv.y, __ATOMIC_RELAXED, __HIP_MEMORY_SCOPE_AGENT);
  __hip_atomic_fetch_add(o + 2, w * hv.z, __ATOMIC_RELAXED, __HIP_MEMORY_SCOPE_AGENT);
  __hip_atomic_fetch_add(o + 3, w * hv.w, __ATOMIC_RELAXED, __HIP_MEMORY_SCOPE_AGENT);
}

__global__ void relu_k(float* __restrict__ a, int n) {
  int t = blockIdx.x * blockDim.x + threadIdx.x;
  if (t < n) a[t] = fmaxf(a[t], 0.0f);
}

// ---------------- launch ----------------
extern "C" void kernel_launch(void* const* d_in, const int* in_sizes, int n_in,
                              void* d_out, int out_size, void* d_ws, size_t ws_size,
                              hipStream_t stream) {
  (void)in_sizes; (void)n_in; (void)out_size; (void)ws_size;
  const float* x  = (const float*)d_in[0];
  const int*   ei = (const int*)d_in[1];
  const float* W1 = (const float*)d_in[2];
  const float* b1 = (const float*)d_in[3];
  const float* W2 = (const float*)d_in[4];
  const float* b2 = (const float*)d_in[5];
  const int* src = ei;            // edge_index[0]
  const int* dst = ei + E_EDGES;  // edge_index[1]
  float* out = (float*)d_out;

  char* ws = (char*)d_ws;
  float* dinv = (float*)ws;  ws += (((size_t)N_NODES * 4) + 255) / 256 * 256;
  float* H1   = (float*)ws;  ws += (size_t)N_NODES * C_HID * 4;
  float* A1   = (float*)ws;  ws += (size_t)N_NODES * C_HID * 4;
  float* H2   = (float*)ws;  // N_NODES * C_OUT floats

  const dim3 B(256);
  // symmetric normalization weights
  deg_init_k<<<(N_NODES + 255) / 256, B, 0, stream>>>(dinv, N_NODES);
  deg_count_k<<<(E_EDGES + 255) / 256, B, 0, stream>>>(dinv, dst, E_EDGES);
  rsqrt_k<<<(N_NODES + 255) / 256, B, 0, stream>>>(dinv, N_NODES);

  // layer 1: H1 = x @ W1 ; A1 = relu(segment_sum(norm*H1[src]) + b1)
  gemm_wmma_k<<<dim3(N_NODES / 16, C_HID / 64), dim3(128), 0, stream>>>(x, W1, H1, C_IN, C_HID);
  init_out_k<C_HID><<<(N_NODES * C_HID + 255) / 256, B, 0, stream>>>(H1, dinv, b1, A1, N_NODES);
  scatter_k<C_HID><<<(E_EDGES * (C_HID / 4) + 255) / 256, B, 0, stream>>>(H1, dinv, src, dst, A1, E_EDGES);
  relu_k<<<(N_NODES * C_HID + 255) / 256, B, 0, stream>>>(A1, N_NODES * C_HID);

  // layer 2: H2 = A1 @ W2 ; out = segment_sum(norm*H2[src]) + b2
  gemm_wmma_k<<<dim3(N_NODES / 16, C_OUT / 64), dim3(128), 0, stream>>>(A1, W2, H2, C_HID, C_OUT);
  init_out_k<C_OUT><<<(N_NODES * C_OUT + 255) / 256, B, 0, stream>>>(H2, dinv, b2, out, N_NODES);
  scatter_k<C_OUT><<<(E_EDGES * (C_OUT / 4) + 255) / 256, B, 0, stream>>>(H2, dinv, src, dst, out, E_EDGES);
}
